// Block_72636486910627
// MI455X (gfx1250) — matrix operand order repaired and checked by reading, not compiled, not run on hardware
//
#include <hip/hip_runtime.h>

typedef __bf16 bf16;
typedef __attribute__((ext_vector_type(16))) __bf16 v16bf;
typedef __attribute__((ext_vector_type(8)))  __bf16 v8bf;
typedef __attribute__((ext_vector_type(8)))  float  v8f;

union BFrag {
  v16bf v;
  v8bf  h[2];
  bf16  e[16];
};

constexpr int Bc = 8, Tc = 2048, Cc = 64, Hc = 4, Dc = 16, Fc = 256;

__device__ static __forceinline__ v8f zero8() {
  v8f z = {0.f, 0.f, 0.f, 0.f, 0.f, 0.f, 0.f, 0.f};
  return z;
}

__device__ static __forceinline__ v8f wmma_bf16(v16bf a, v16bf b, v8f c) {
  // D = A(16x32 bf16) * B(32x16 bf16) + C(16x16 f32)
  return __builtin_amdgcn_wmma_f32_16x16x32_bf16(false, a, false, b, (short)0, c,
                                                 false, false);
}

__device__ static __forceinline__ float half16_max(float v) {
  #pragma unroll
  for (int m = 1; m <= 8; m <<= 1) v = fmaxf(v, __shfl_xor(v, m, 32));
  return v;
}
__device__ static __forceinline__ float half16_sum(float v) {
  #pragma unroll
  for (int m = 1; m <= 8; m <<= 1) v += __shfl_xor(v, m, 32);
  return v;
}

// Load one pre-swizzled B-fragment: 32 contiguous bytes per lane.
__device__ static __forceinline__ BFrag load_pfrag(const bf16* __restrict__ WB,
                                                   int fragIdx, int lane) {
  const bf16* p = WB + ((size_t)fragIdx * 32 + lane) * 16;
  BFrag f;
  f.h[0] = *(const v8bf*)p;
  f.h[1] = *(const v8bf*)(p + 8);
  return f;
}

// ---------------------------------------------------------------------------
// Weight pre-swizzle: fp32 W[batch][K x N] (row-major) -> bf16 B-fragments.
// One 32-lane block per fragment.  Fragment order within a matrix: j-major
// (K-chunk of 32), then n-tile of 16.  Batches are concatenated.
// packed[(frag*32 + lane)*16 + i] = W[(32j + (lane>>4)*8 + (i<8 ? i : i+8))*N + nt*16 + (lane&15)]
// ---------------------------------------------------------------------------
__global__ __launch_bounds__(32) void k_pack(const float* __restrict__ W,
                                             int K, int N, int batchStride,
                                             bf16* __restrict__ out) {
  const int lane = threadIdx.x;
  const int nn = N >> 4, nk = K >> 5;
  const int fragsPerMat = nk * nn;
  int f = blockIdx.x;
  const int bm = f / fragsPerMat;
  f -= bm * fragsPerMat;
  const int j = f / nn;
  const int nt = f - j * nn;

  const float* Wm = W + (size_t)bm * batchStride;
  const int krow0 = 32 * j + (lane >> 4) * 8;
  const int ncol = nt * 16 + (lane & 15);

  BFrag fr;
  #pragma unroll
  for (int i = 0; i < 16; ++i)
    fr.e[i] = (bf16)Wm[(size_t)(krow0 + (i < 8 ? i : i + 8)) * N + ncol];

  bf16* dst = out + ((size_t)blockIdx.x * 32 + lane) * 16;
  *(v8bf*)dst = fr.h[0];
  *(v8bf*)(dst + 8) = fr.h[1];
}

// ---------------------------------------------------------------------------
// Kernel 1: LN1 + QKV projection.  One wave per (b, 16-row tile).
// Q,K stored bf16 [B,H,T,D]; V stored transposed bf16 [B,H,D,T].
// ---------------------------------------------------------------------------
__global__ __launch_bounds__(32) void k_ln_qkv(
    const float* __restrict__ x,
    const bf16* __restrict__ WqB, const bf16* __restrict__ WkB,
    const bf16* __restrict__ WvB,
    const float* __restrict__ gamma1, const float* __restrict__ beta1,
    bf16* __restrict__ Qw, bf16* __restrict__ Kw, bf16* __restrict__ Vt) {
  __shared__ bf16 hbuf[16 * 64];

  const int lane = threadIdx.x;
  const int r = lane & 15;
  const int half = lane >> 4;
  const int bid = blockIdx.x;
  const int b = bid >> 7;              // T/16 = 128
  const int t0 = (bid & 127) << 4;

  // ---- LayerNorm: lane handles row (t0+r), cols [32*half, 32*half+32) ----
  const float* xrow = x + ((size_t)b * Tc + t0 + r) * Cc + half * 32;
  float vbuf[32], gb[32], bb[32];
  float s = 0.f, s2 = 0.f;
  #pragma unroll
  for (int j = 0; j < 8; ++j) {
    float4 t = ((const float4*)xrow)[j];
    float4 g = ((const float4*)(gamma1 + half * 32))[j];
    float4 be = ((const float4*)(beta1 + half * 32))[j];
    vbuf[4 * j + 0] = t.x; vbuf[4 * j + 1] = t.y;
    vbuf[4 * j + 2] = t.z; vbuf[4 * j + 3] = t.w;
    gb[4 * j + 0] = g.x; gb[4 * j + 1] = g.y; gb[4 * j + 2] = g.z; gb[4 * j + 3] = g.w;
    bb[4 * j + 0] = be.x; bb[4 * j + 1] = be.y; bb[4 * j + 2] = be.z; bb[4 * j + 3] = be.w;
    s += t.x + t.y + t.z + t.w;
    s2 += t.x * t.x + t.y * t.y + t.z * t.z + t.w * t.w;
  }
  s += __shfl_xor(s, 16, 32);
  s2 += __shfl_xor(s2, 16, 32);
  const float mu = s * (1.f / 64.f);
  const float var = s2 * (1.f / 64.f) - mu * mu;
  const float rstd = rsqrtf(var + 1e-5f);
  #pragma unroll
  for (int j = 0; j < 32; ++j) {
    float hv = (vbuf[j] - mu) * rstd * gb[j] + bb[j];
    hbuf[r * 64 + half * 32 + j] = (bf16)hv;
  }
  __syncthreads();

  // ---- A-fragments of h (two K=32 chunks over C=64) ----
  const int akb = half * 8;
  BFrag a0, a1;
  a0.h[0] = *(const v8bf*)(&hbuf[r * 64 + akb]);
  a0.h[1] = *(const v8bf*)(&hbuf[r * 64 + 16 + akb]);
  a1.h[0] = *(const v8bf*)(&hbuf[r * 64 + 32 + akb]);
  a1.h[1] = *(const v8bf*)(&hbuf[r * 64 + 48 + akb]);

  #pragma unroll
  for (int hd = 0; hd < Hc; ++hd) {
    // Per head: nk=2, nn=1 -> frags (hd*2 + j)
    BFrag bq0 = load_pfrag(WqB, hd * 2 + 0, lane), bq1 = load_pfrag(WqB, hd * 2 + 1, lane);
    BFrag bk0 = load_pfrag(WkB, hd * 2 + 0, lane), bk1 = load_pfrag(WkB, hd * 2 + 1, lane);
    BFrag bv0 = load_pfrag(WvB, hd * 2 + 0, lane), bv1 = load_pfrag(WvB, hd * 2 + 1, lane);

    v8f aq = zero8(), ak = zero8(), av = zero8();
    aq = wmma_bf16(a0.v, bq0.v, aq); aq = wmma_bf16(a1.v, bq1.v, aq);
    ak = wmma_bf16(a0.v, bk0.v, ak); ak = wmma_bf16(a1.v, bk1.v, ak);
    av = wmma_bf16(a0.v, bv0.v, av); av = wmma_bf16(a1.v, bv1.v, av);

    const size_t bh = (size_t)b * Hc + hd;
    #pragma unroll
    for (int g = 0; g < 8; ++g) {
      int trow = t0 + half * 8 + g;
      Qw[(bh * Tc + trow) * Dc + r] = (bf16)aq[g];
      Kw[(bh * Tc + trow) * Dc + r] = (bf16)ak[g];
      Vt[(bh * Dc + r) * Tc + trow] = (bf16)av[g];
    }
  }
}

// ---------------------------------------------------------------------------
// Kernel 2: causal flash attention.  One wave per (b, h, 16-row q tile).
// ---------------------------------------------------------------------------
__global__ __launch_bounds__(32) void k_attn(
    const bf16* __restrict__ Qw, const bf16* __restrict__ Kw,
    const bf16* __restrict__ Vt, bf16* __restrict__ AO) {
  __shared__ bf16 pbuf[16 * 32];

  const int lane = threadIdx.x;
  const int r = lane & 15;
  const int half = lane >> 4;
  const int bid = blockIdx.x;
  const int bh = bid >> 7;
  const int t0 = (bid & 127) << 4;
  const int b = bh >> 2, hd = bh & 3;

  // Q A-fragment: K = d(16), padded to 32 with zeros
  BFrag qa;
  qa.h[0] = *(const v8bf*)(Qw + ((size_t)bh * Tc + t0 + r) * Dc);
  qa.h[1] = *(const v8bf*)(Qw + ((size_t)bh * Tc + t0 + r) * Dc + 8);
  if (half) { for (int i = 0; i < 16; ++i) qa.e[i] = (bf16)0.f; }

  float mrow[8], lrow[8];
  #pragma unroll
  for (int g = 0; g < 8; ++g) { mrow[g] = -INFINITY; lrow[g] = 0.f; }
  v8f oacc = zero8();

  const float scale = 0.25f;  // 1/sqrt(16)
  const int rowbase = t0 + half * 8;

  for (int s0 = 0; s0 < t0 + 16; s0 += 32) {
    // K^T B-fragments (rows 16..31 are the zero d-padding -> lanes>=16 zero)
    BFrag kb0, kb1;
    if (half == 0) {
      const bf16* k0 = Kw + ((size_t)bh * Tc + s0 + r) * Dc;
      const bf16* k1 = Kw + ((size_t)bh * Tc + s0 + 16 + r) * Dc;
      kb0.h[0] = *(const v8bf*)k0; kb0.h[1] = *(const v8bf*)(k0 + 8);
      kb1.h[0] = *(const v8bf*)k1; kb1.h[1] = *(const v8bf*)(k1 + 8);
    } else {
      #pragma unroll
      for (int i = 0; i < 16; ++i) { kb0.e[i] = (bf16)0.f; kb1.e[i] = (bf16)0.f; }
    }

    v8f p0 = wmma_bf16(qa.v, kb0.v, zero8());
    v8f p1 = wmma_bf16(qa.v, kb1.v, zero8());

    const bool need_mask = (s0 + 31 > t0);
    #pragma unroll
    for (int g = 0; g < 8; ++g) {
      float pa = p0[g] * scale, pb = p1[g] * scale;
      if (need_mask) {
        if (s0 + r > rowbase + g) pa = -INFINITY;
        if (s0 + 16 + r > rowbase + g) pb = -INFINITY;
      }
      float rm = half16_max(fmaxf(pa, pb));
      float nm = fmaxf(mrow[g], rm);
      float alpha = __expf(mrow[g] - nm);
      float e0 = __expf(pa - nm), e1 = __expf(pb - nm);
      float rs = half16_sum(e0 + e1);
      lrow[g] = lrow[g] * alpha + rs;
      mrow[g] = nm;
      oacc[g] *= alpha;
      p0[g] = e0; p1[g] = e1;
    }

    // Transpose P (C/D layout -> A layout) through LDS
    __syncthreads();  // protect previous iteration's pbuf reads
    #pragma unroll
    for (int g = 0; g < 8; ++g) {
      pbuf[(half * 8 + g) * 32 + r] = (bf16)p0[g];
      pbuf[(half * 8 + g) * 32 + 16 + r] = (bf16)p1[g];
    }
    __syncthreads();

    BFrag pfrag;
    pfrag.h[0] = *(const v8bf*)(&pbuf[r * 32 + half * 8]);
    pfrag.h[1] = *(const v8bf*)(&pbuf[r * 32 + 16 + half * 8]);

    // V B-fragment: lane = column d, s ranges 8h..8h+7 and 16+8h.. from V^T
    BFrag vb;
    const bf16* vrow = Vt + ((size_t)bh * Dc + r) * Tc + s0 + half * 8;
    vb.h[0] = *(const v8bf*)vrow;
    vb.h[1] = *(const v8bf*)(vrow + 16);

    oacc = wmma_bf16(pfrag.v, vb.v, oacc);
  }

  #pragma unroll
  for (int g = 0; g < 8; ++g) {
    float o = oacc[g] / lrow[g];
    int trow = rowbase + g;
    AO[((size_t)b * Tc + trow) * Cc + hd * 16 + r] = (bf16)o;
  }
}

// ---------------------------------------------------------------------------
// Kernel 3: out = y + relu(LN2(y) @ W1 + b1) @ W2 + b2,
//           y = x + attn_out @ Wo + bo.   One wave per (b, 16-row tile).
// ---------------------------------------------------------------------------
__global__ __launch_bounds__(32) void k_proj_mlp(
    const float* __restrict__ x, const bf16* __restrict__ AO,
    const bf16* __restrict__ WoB, const float* __restrict__ bo,
    const bf16* __restrict__ W1B, const float* __restrict__ b1,
    const bf16* __restrict__ W2B, const float* __restrict__ b2,
    const float* __restrict__ gamma2, const float* __restrict__ beta2,
    float* __restrict__ out) {
  __shared__ bf16 hbuf[16 * 64];
  __shared__ bf16 zbuf[16 * 256];

  const int lane = threadIdx.x;
  const int r = lane & 15;
  const int half = lane >> 4;
  const int bid = blockIdx.x;
  const int b = bid >> 7;
  const int t0 = (bid & 127) << 4;
  const int akb = half * 8;

  // ---- attn-out projection + residual: y = x + AO @ Wo + bo ----
  BFrag a0, a1;
  const bf16* ar = AO + ((size_t)b * Tc + t0 + r) * Cc;
  a0.h[0] = *(const v8bf*)(ar + akb);
  a0.h[1] = *(const v8bf*)(ar + 16 + akb);
  a1.h[0] = *(const v8bf*)(ar + 32 + akb);
  a1.h[1] = *(const v8bf*)(ar + 48 + akb);

  float y[4][8];
  #pragma unroll
  for (int nt = 0; nt < 4; ++nt) {
    // Wo: nk=2, nn=4 -> frag (j*4 + nt)
    BFrag w0 = load_pfrag(WoB, 0 * 4 + nt, lane);
    BFrag w1 = load_pfrag(WoB, 1 * 4 + nt, lane);
    v8f acc = zero8();
    acc = wmma_bf16(a0.v, w0.v, acc);
    acc = wmma_bf16(a1.v, w1.v, acc);
    const int col = nt * 16 + r;
    const float bov = bo[col];
    #pragma unroll
    for (int g = 0; g < 8; ++g) {
      int trow = t0 + half * 8 + g;
      y[nt][g] = x[((size_t)b * Tc + trow) * Cc + col] + acc[g] + bov;
    }
  }

  // ---- LN2 ----
  #pragma unroll
  for (int g = 0; g < 8; ++g) {
    float ss = y[0][g] + y[1][g] + y[2][g] + y[3][g];
    float qq = y[0][g] * y[0][g] + y[1][g] * y[1][g] +
               y[2][g] * y[2][g] + y[3][g] * y[3][g];
    ss = half16_sum(ss);
    qq = half16_sum(qq);
    float mu = ss * (1.f / 64.f);
    float var = qq * (1.f / 64.f) - mu * mu;
    float rstd = rsqrtf(var + 1e-5f);
    #pragma unroll
    for (int nt = 0; nt < 4; ++nt) {
      int col = nt * 16 + r;
      float hv = (y[nt][g] - mu) * rstd * gamma2[col] + beta2[col];
      hbuf[(half * 8 + g) * 64 + col] = (bf16)hv;
    }
  }
  __syncthreads();

  BFrag ha0, ha1;
  ha0.h[0] = *(const v8bf*)(&hbuf[r * 64 + akb]);
  ha0.h[1] = *(const v8bf*)(&hbuf[r * 64 + 16 + akb]);
  ha1.h[0] = *(const v8bf*)(&hbuf[r * 64 + 32 + akb]);
  ha1.h[1] = *(const v8bf*)(&hbuf[r * 64 + 48 + akb]);

  // ---- MLP layer 1 + ReLU -> zbuf (bf16, 16 x 256) ----
  #pragma unroll
  for (int n1 = 0; n1 < 16; ++n1) {
    // W1: nk=2, nn=16 -> frag (j*16 + n1)
    BFrag w0 = load_pfrag(W1B, 0 * 16 + n1, lane);
    BFrag w1 = load_pfrag(W1B, 1 * 16 + n1, lane);
    v8f acc = zero8();
    acc = wmma_bf16(ha0.v, w0.v, acc);
    acc = wmma_bf16(ha1.v, w1.v, acc);
    const int col = n1 * 16 + r;
    const float b1v = b1[col];
    #pragma unroll
    for (int g = 0; g < 8; ++g) {
      float z = fmaxf(acc[g] + b1v, 0.f);
      zbuf[(half * 8 + g) * 256 + col] = (bf16)z;
    }
  }
  __syncthreads();

  // ---- MLP layer 2 + residual ----
  v8f acc2[4];
  #pragma unroll
  for (int n2 = 0; n2 < 4; ++n2) acc2[n2] = zero8();
  #pragma unroll
  for (int kc = 0; kc < 8; ++kc) {
    BFrag pf;
    pf.h[0] = *(const v8bf*)(&zbuf[r * 256 + kc * 32 + akb]);
    pf.h[1] = *(const v8bf*)(&zbuf[r * 256 + kc * 32 + 16 + akb]);
    #pragma unroll
    for (int n2 = 0; n2 < 4; ++n2) {
      // W2: nk=8, nn=4 -> frag (kc*4 + n2)
      BFrag w = load_pfrag(W2B, kc * 4 + n2, lane);
      acc2[n2] = wmma_bf16(pf.v, w.v, acc2[n2]);
    }
  }
  #pragma unroll
  for (int n2 = 0; n2 < 4; ++n2) {
    const int col = n2 * 16 + r;
    const float b2v = b2[col];
    #pragma unroll
    for (int g = 0; g < 8; ++g) {
      int trow = t0 + half * 8 + g;
      out[((size_t)b * Tc + trow) * Cc + col] = y[n2][g] + acc2[n2][g] + b2v;
    }
  }
}

// ---------------------------------------------------------------------------
extern "C" void kernel_launch(void* const* d_in, const int* in_sizes, int n_in,
                              void* d_out, int out_size, void* d_ws, size_t ws_size,
                              hipStream_t stream) {
  (void)in_sizes; (void)n_in; (void)out_size; (void)ws_size;
  const float* x   = (const float*)d_in[0];
  const float* Wq  = (const float*)d_in[1];
  const float* Wk  = (const float*)d_in[2];
  const float* Wv  = (const float*)d_in[3];
  const float* Wo  = (const float*)d_in[4];
  const float* bo  = (const float*)d_in[5];
  const float* W1  = (const float*)d_in[6];
  const float* b1  = (const float*)d_in[7];
  const float* W2  = (const float*)d_in[8];
  const float* b2  = (const float*)d_in[9];
  const float* g1  = (const float*)d_in[10];
  const float* be1 = (const float*)d_in[11];
  const float* g2  = (const float*)d_in[12];
  const float* be2 = (const float*)d_in[13];

  const size_t qkv_elems = (size_t)Bc * Hc * Tc * Dc;   // 1M bf16 each
  bf16* Qw = (bf16*)d_ws;
  bf16* Kw = Qw + qkv_elems;
  bf16* Vt = Kw + qkv_elems;
  bf16* AO = Vt + qkv_elems;                            // [B,T,C] bf16

  // Pre-swizzled weight fragments (512 bf16 per fragment)
  bf16* WqB = AO + (size_t)Bc * Tc * Cc;                // H*2 frags = 4096
  bf16* WkB = WqB + (size_t)Hc * 2 * 512;
  bf16* WvB = WkB + (size_t)Hc * 2 * 512;
  bf16* WoB = WvB + (size_t)Hc * 2 * 512;               // 8 frags  = 4096
  bf16* W1B = WoB + (size_t)8 * 512;                    // 32 frags = 16384
  bf16* W2B = W1B + (size_t)32 * 512;                   // 32 frags = 16384

  dim3 blk(32);
  k_pack<<<Hc * 2, blk, 0, stream>>>(Wq, Cc, Dc, Cc * Dc, WqB);
  k_pack<<<Hc * 2, blk, 0, stream>>>(Wk, Cc, Dc, Cc * Dc, WkB);
  k_pack<<<Hc * 2, blk, 0, stream>>>(Wv, Cc, Dc, Cc * Dc, WvB);
  k_pack<<<8,      blk, 0, stream>>>(Wo, Cc, Cc, Cc * Cc, WoB);
  k_pack<<<32,     blk, 0, stream>>>(W1, Cc, Fc, Cc * Fc, W1B);
  k_pack<<<32,     blk, 0, stream>>>(W2, Fc, Cc, Fc * Cc, W2B);

  k_ln_qkv<<<Bc * Tc / 16, blk, 0, stream>>>(x, WqB, WkB, WvB, g1, be1, Qw, Kw, Vt);
  k_attn<<<Bc * Hc * Tc / 16, blk, 0, stream>>>(Qw, Kw, Vt, AO);
  k_proj_mlp<<<Bc * Tc / 16, blk, 0, stream>>>(x, AO, WoB, bo, W1B, b1, W2B, b2,
                                               g2, be2, (float*)d_out);
}
